// DiagSSMBlock_31387620999391
// MI455X (gfx1250) — matrix operand order repaired and checked by reading, not compiled, not run on hardware
//
#include <hip/hip_runtime.h>
#include <hip/hip_bf16.h>
#include <math.h>

// ---- CDNA5 vector types (match probe-confirmed WMMA builtin signatures) ----
typedef __attribute__((ext_vector_type(16))) __bf16 v16bf;
typedef __attribute__((ext_vector_type(8)))  __bf16 v8bf;
typedef __attribute__((ext_vector_type(8)))  float  v8f;

#define T_DIM 4096
#define H_DIM 2048
#define APITCH 40   // bf16 elems per LDS row: 32 + 8 pad -> 80B rows (16B aligned)

// f32 -> bf16 (round to nearest even)
__device__ __forceinline__ unsigned short bf16_rne(float f) {
    unsigned int u = __float_as_uint(f);
    u += 0x7fffu + ((u >> 16) & 1u);
    return (unsigned short)(u >> 16);
}
__device__ __forceinline__ float bf16_up(unsigned short h) {
    return __uint_as_float(((unsigned int)h) << 16);
}

// ---------------------------------------------------------------------------
// Prep kernel A: elementwise split f32 -> (hi, lo) bf16, same layout.
// ---------------------------------------------------------------------------
__global__ __launch_bounds__(256) void split_bf16(
    const float* __restrict__ in, unsigned short* __restrict__ hi,
    unsigned short* __restrict__ lo)
{
    size_t i = (size_t)blockIdx.x * 256 + threadIdx.x;       // float4 index
    float4 v = ((const float4*)in)[i];
    float vv[4] = {v.x, v.y, v.z, v.w};
    unsigned short hs[4], ls[4];
    #pragma unroll
    for (int j = 0; j < 4; ++j) {
        hs[j] = bf16_rne(vv[j]);
        ls[j] = bf16_rne(vv[j] - bf16_up(hs[j]));
    }
    uint2 hp, lp;
    hp.x = (unsigned)hs[0] | ((unsigned)hs[1] << 16);
    hp.y = (unsigned)hs[2] | ((unsigned)hs[3] << 16);
    lp.x = (unsigned)ls[0] | ((unsigned)ls[1] << 16);
    lp.y = (unsigned)ls[2] | ((unsigned)ls[3] << 16);
    ((uint2*)hi)[i] = hp;
    ((uint2*)lo)[i] = lp;
}

// ---------------------------------------------------------------------------
// Prep kernel B: split + transpose B[K][N] f32 -> Bt[N][K] (hi, lo) bf16.
// 64x64 tiles through LDS; coalesced in, coalesced out.
// ---------------------------------------------------------------------------
__global__ __launch_bounds__(256) void split_transpose_bf16(
    const float* __restrict__ B, unsigned short* __restrict__ bthi,
    unsigned short* __restrict__ btlo)
{
    __shared__ __align__(16) unsigned short Th[64 * 72];
    __shared__ __align__(16) unsigned short Tl[64 * 72];
    const int k0 = blockIdx.y * 64;
    const int n0 = blockIdx.x * 64;
    const int tid = threadIdx.x;

    #pragma unroll
    for (int r = 0; r < 4; ++r) {
        int idx = tid + r * 256;             // 0..1023 over 64 rows x 16 float4
        int row = idx >> 4;                  // k within tile
        int cv  = (idx & 15) << 2;           // n within tile
        float4 v = *(const float4*)(B + (size_t)(k0 + row) * H_DIM + n0 + cv);
        float vv[4] = {v.x, v.y, v.z, v.w};
        #pragma unroll
        for (int j = 0; j < 4; ++j) {
            unsigned short h = bf16_rne(vv[j]);
            Th[(cv + j) * 72 + row] = h;
            Tl[(cv + j) * 72 + row] = bf16_rne(vv[j] - bf16_up(h));
        }
    }
    __syncthreads();
    #pragma unroll
    for (int r = 0; r < 4; ++r) {
        int idx = tid + r * 256;             // 0..1023
        int arr = idx >> 9;                  // 0 = hi, 1 = lo
        int c   = idx & 511;                 // 512 chunks of 8 bf16
        int nrow = c >> 3;
        int ch   = (c & 7) << 3;
        const unsigned short* src = arr ? Tl : Th;
        unsigned short* dst = arr ? btlo : bthi;
        uint4 d = *(const uint4*)&src[nrow * 72 + ch];
        *(uint4*)(dst + (size_t)(n0 + nrow) * H_DIM + k0 + ch) = d;
    }
}

// ---------------------------------------------------------------------------
// Main GEMM (preconverted): S = X @ B via bf16x3 WMMA.
// 128x128 block, 8 waves, each wave 32x64 (2x4 tiles) -> 24 WMMAs / K-step.
// Staging is pure b128 copies (no VALU conversion in the hot loop).
// ---------------------------------------------------------------------------
__global__ __launch_bounds__(256) void ssm_gemm_wmma(
    const unsigned short* __restrict__ Xhi, const unsigned short* __restrict__ Xlo,
    const unsigned short* __restrict__ Bthi, const unsigned short* __restrict__ Btlo,
    float* __restrict__ S)
{
    __shared__ __align__(16) unsigned short Ahi[128 * APITCH];
    __shared__ __align__(16) unsigned short Alo[128 * APITCH];
    __shared__ __align__(16) unsigned short Bhi[128 * APITCH];
    __shared__ __align__(16) unsigned short Blo[128 * APITCH];

    const int tid   = threadIdx.x;
    const int lane  = tid & 31;
    const int wave  = tid >> 5;          // 0..7
    const int waveM = wave >> 1;         // 0..3  (rows of 32)
    const int waveN = wave & 1;          // 0..1  (cols of 64)
    const int m0 = blockIdx.y * 128;
    const int n0 = blockIdx.x * 128;

    v8f acc[2][4];
    #pragma unroll
    for (int mi = 0; mi < 2; ++mi)
        #pragma unroll
        for (int ni = 0; ni < 4; ++ni)
            acc[mi][ni] = (v8f)(0.0f);

    const int rsel = lane & 15;
    const int kbA  = (lane < 16) ? 0 : 8;    // A frag: K halves interleave
    const int kbB  = (lane < 16) ? 0 : 8;    // B frag: K halves interleave

    for (int k0 = 0; k0 < H_DIM; k0 += 32) {
        // ---- stage A: 128 rows x 32 bf16 x {hi,lo} = 1024 x 16B chunks ----
        #pragma unroll
        for (int r = 0; r < 4; ++r) {
            int idx = tid + r * 256;
            int arr = idx >> 9;
            int c   = idx & 511;             // 128 rows x 4 chunks
            int row = c >> 2;
            int ch  = (c & 3) << 3;
            const unsigned short* src = arr ? Xlo : Xhi;
            unsigned short* dst = arr ? Alo : Ahi;
            uint4 d = *(const uint4*)(src + (size_t)(m0 + row) * H_DIM + k0 + ch);
            *(uint4*)&dst[row * APITCH + ch] = d;
        }
        // ---- stage B (already [N][K]): identical copy pattern ----
        #pragma unroll
        for (int r = 0; r < 4; ++r) {
            int idx = tid + r * 256;
            int arr = idx >> 9;
            int c   = idx & 511;
            int row = c >> 2;                // n within tile
            int ch  = (c & 3) << 3;
            const unsigned short* src = arr ? Btlo : Bthi;
            unsigned short* dst = arr ? Blo : Bhi;
            uint4 d = *(const uint4*)(src + (size_t)(n0 + row) * H_DIM + k0 + ch);
            *(uint4*)&dst[row * APITCH + ch] = d;
        }
        __syncthreads();

        // ---- fragments ----
        v16bf ah[2], al[2], bh[4], bl[4];
        #pragma unroll
        for (int mi = 0; mi < 2; ++mi) {
            int row = waveM * 32 + mi * 16 + rsel;
            const v8bf a0 = *(const v8bf*)&Ahi[row * APITCH + kbA];
            const v8bf a1 = *(const v8bf*)&Ahi[row * APITCH + kbA + 16];
            const v8bf a2 = *(const v8bf*)&Alo[row * APITCH + kbA];
            const v8bf a3 = *(const v8bf*)&Alo[row * APITCH + kbA + 16];
            #pragma unroll
            for (int j = 0; j < 8; ++j) {
                ah[mi][j] = a0[j]; ah[mi][j + 8] = a1[j];
                al[mi][j] = a2[j]; al[mi][j + 8] = a3[j];
            }
        }
        #pragma unroll
        for (int ni = 0; ni < 4; ++ni) {
            int col = waveN * 64 + ni * 16 + rsel;
            const v8bf b0 = *(const v8bf*)&Bhi[col * APITCH + kbB];
            const v8bf b1 = *(const v8bf*)&Bhi[col * APITCH + kbB + 16];
            const v8bf b2 = *(const v8bf*)&Blo[col * APITCH + kbB];
            const v8bf b3 = *(const v8bf*)&Blo[col * APITCH + kbB + 16];
            #pragma unroll
            for (int j = 0; j < 8; ++j) {
                bh[ni][j] = b0[j]; bh[ni][j + 8] = b1[j];
                bl[ni][j] = b2[j]; bl[ni][j + 8] = b3[j];
            }
        }

        // ---- 24 WMMAs: 3 split-precision passes x 8 tiles ----
        #pragma unroll
        for (int mi = 0; mi < 2; ++mi)
            #pragma unroll
            for (int ni = 0; ni < 4; ++ni)
                acc[mi][ni] = __builtin_amdgcn_wmma_f32_16x16x32_bf16(
                    false, ah[mi], false, bh[ni], (short)0, acc[mi][ni], false, false);
        #pragma unroll
        for (int mi = 0; mi < 2; ++mi)
            #pragma unroll
            for (int ni = 0; ni < 4; ++ni)
                acc[mi][ni] = __builtin_amdgcn_wmma_f32_16x16x32_bf16(
                    false, ah[mi], false, bl[ni], (short)0, acc[mi][ni], false, false);
        #pragma unroll
        for (int mi = 0; mi < 2; ++mi)
            #pragma unroll
            for (int ni = 0; ni < 4; ++ni)
                acc[mi][ni] = __builtin_amdgcn_wmma_f32_16x16x32_bf16(
                    false, al[mi], false, bh[ni], (short)0, acc[mi][ni], false, false);

        __syncthreads();
    }

    // ---- store: VGPR r holds M=r (lanes 0-15) / M=8+r (lanes 16-31) ----
    const int rofs = (lane < 16) ? 0 : 8;
    #pragma unroll
    for (int mi = 0; mi < 2; ++mi) {
        #pragma unroll
        for (int ni = 0; ni < 4; ++ni) {
            int gm = m0 + waveM * 32 + mi * 16 + rofs;
            int gn = n0 + waveN * 64 + ni * 16 + rsel;
            #pragma unroll
            for (int r = 0; r < 8; ++r)
                S[(size_t)(gm + r) * H_DIM + gn] = acc[mi][ni][r];
        }
    }
}

// ---------------------------------------------------------------------------
// Fallback GEMM (round-1 fused conversion, 64x64/4-wave) if ws is small.
// ---------------------------------------------------------------------------
__global__ __launch_bounds__(128) void ssm_gemm_fused(
    const float* __restrict__ X, const float* __restrict__ Bm, float* __restrict__ S)
{
    __shared__ __align__(16) unsigned short Ahi[64 * APITCH];
    __shared__ __align__(16) unsigned short Alo[64 * APITCH];
    __shared__ __align__(16) unsigned short Bhi[64 * APITCH];
    __shared__ __align__(16) unsigned short Blo[64 * APITCH];

    const int tid = threadIdx.x, lane = tid & 31, wave = tid >> 5;
    const int waveM = wave >> 1, waveN = wave & 1;
    const int m0 = blockIdx.y * 64, n0 = blockIdx.x * 64;

    v8f acc[2][2];
    #pragma unroll
    for (int mi = 0; mi < 2; ++mi)
        #pragma unroll
        for (int ni = 0; ni < 2; ++ni) acc[mi][ni] = (v8f)(0.0f);

    const int rsel = lane & 15;
    const int kbA = (lane < 16) ? 0 : 8;
    const int kbB = (lane < 16) ? 0 : 8;

    for (int k0 = 0; k0 < H_DIM; k0 += 32) {
        #pragma unroll
        for (int r = 0; r < 4; ++r) {
            int idx = tid + r * 128, row = idx >> 3, kv = (idx & 7) << 2;
            float4 v = *(const float4*)(X + (size_t)(m0 + row) * H_DIM + k0 + kv);
            float vv[4] = {v.x, v.y, v.z, v.w};
            #pragma unroll
            for (int j = 0; j < 4; ++j) {
                unsigned short h = bf16_rne(vv[j]);
                Ahi[row * APITCH + kv + j] = h;
                Alo[row * APITCH + kv + j] = bf16_rne(vv[j] - bf16_up(h));
            }
        }
        #pragma unroll
        for (int r = 0; r < 4; ++r) {
            int idx = tid + r * 128, krow = idx >> 4, nv = (idx & 15) << 2;
            float4 v = *(const float4*)(Bm + (size_t)(k0 + krow) * H_DIM + n0 + nv);
            float vv[4] = {v.x, v.y, v.z, v.w};
            #pragma unroll
            for (int j = 0; j < 4; ++j) {
                unsigned short h = bf16_rne(vv[j]);
                Bhi[(nv + j) * APITCH + krow] = h;
                Blo[(nv + j) * APITCH + krow] = bf16_rne(vv[j] - bf16_up(h));
            }
        }
        __syncthreads();

        v16bf ah[2], al[2], bh[2], bl[2];
        #pragma unroll
        for (int mi = 0; mi < 2; ++mi) {
            int row = waveM * 32 + mi * 16 + rsel;
            const v8bf a0 = *(const v8bf*)&Ahi[row * APITCH + kbA];
            const v8bf a1 = *(const v8bf*)&Ahi[row * APITCH + kbA + 16];
            const v8bf a2 = *(const v8bf*)&Alo[row * APITCH + kbA];
            const v8bf a3 = *(const v8bf*)&Alo[row * APITCH + kbA + 16];
            #pragma unroll
            for (int j = 0; j < 8; ++j) {
                ah[mi][j] = a0[j]; ah[mi][j + 8] = a1[j];
                al[mi][j] = a2[j]; al[mi][j + 8] = a3[j];
            }
        }
        #pragma unroll
        for (int ni = 0; ni < 2; ++ni) {
            int col = waveN * 32 + ni * 16 + rsel;
            const v8bf b0 = *(const v8bf*)&Bhi[col * APITCH + kbB];
            const v8bf b1 = *(const v8bf*)&Bhi[col * APITCH + kbB + 16];
            const v8bf b2 = *(const v8bf*)&Blo[col * APITCH + kbB];
            const v8bf b3 = *(const v8bf*)&Blo[col * APITCH + kbB + 16];
            #pragma unroll
            for (int j = 0; j < 8; ++j) {
                bh[ni][j] = b0[j]; bh[ni][j + 8] = b1[j];
                bl[ni][j] = b2[j]; bl[ni][j + 8] = b3[j];
            }
        }
        #pragma unroll
        for (int mi = 0; mi < 2; ++mi)
            #pragma unroll
            for (int ni = 0; ni < 2; ++ni)
                acc[mi][ni] = __builtin_amdgcn_wmma_f32_16x16x32_bf16(
                    false, ah[mi], false, bh[ni], (short)0, acc[mi][ni], false, false);
        #pragma unroll
        for (int mi = 0; mi < 2; ++mi)
            #pragma unroll
            for (int ni = 0; ni < 2; ++ni)
                acc[mi][ni] = __builtin_amdgcn_wmma_f32_16x16x32_bf16(
                    false, ah[mi], false, bl[ni], (short)0, acc[mi][ni], false, false);
        #pragma unroll
        for (int mi = 0; mi < 2; ++mi)
            #pragma unroll
            for (int ni = 0; ni < 2; ++ni)
                acc[mi][ni] = __builtin_amdgcn_wmma_f32_16x16x32_bf16(
                    false, al[mi], false, bh[ni], (short)0, acc[mi][ni], false, false);
        __syncthreads();
    }

    const int rofs = (lane < 16) ? 0 : 8;
    #pragma unroll
    for (int mi = 0; mi < 2; ++mi)
        #pragma unroll
        for (int ni = 0; ni < 2; ++ni) {
            int gm = m0 + waveM * 32 + mi * 16 + rofs;
            int gn = n0 + waveN * 32 + ni * 16 + rsel;
            #pragma unroll
            for (int r = 0; r < 8; ++r)
                S[(size_t)(gm + r) * H_DIM + gn] = acc[mi][ni][r];
        }
}

// ---------------------------------------------------------------------------
// Scan: diagonal tanh recurrence, one thread per channel.
// ---------------------------------------------------------------------------
__global__ __launch_bounds__(256) void ssm_scan(
    const float* __restrict__ a, const float* __restrict__ s, float* __restrict__ out)
{
    const int i = blockIdx.x * blockDim.x + threadIdx.x;
    const float av = a[i];
    const float* sp = s + i;
    float* op = out + i;

    float h = 0.0f;
    float cur = sp[0];
    for (int t = 0; t < T_DIM; ++t) {
        int tn = (t + 1 < T_DIM) ? (t + 1) : (T_DIM - 1);
        float nxt = sp[(size_t)tn * H_DIM];
        float z = fmaf(av, h, cur);
#if __has_builtin(__builtin_amdgcn_tanhf)
        h = __builtin_amdgcn_tanhf(z);                     // v_tanh_f32 on gfx1250
#else
        h = tanhf(z);
#endif
        op[(size_t)t * H_DIM] = h;
        cur = nxt;
    }
}

extern "C" void kernel_launch(void* const* d_in, const int* in_sizes, int n_in,
                              void* d_out, int out_size, void* d_ws, size_t ws_size,
                              hipStream_t stream) {
    (void)in_sizes; (void)n_in; (void)out_size;
    const float* x = (const float*)d_in[0];   // (T, H)
    const float* a = (const float*)d_in[1];   // (H,)
    const float* b = (const float*)d_in[2];   // (H, H)
    float* out = (float*)d_out;               // (T, H)

    const size_t S_BYTES  = (size_t)T_DIM * H_DIM * 4;   // 33.5 MB
    const size_t XH_ELEMS = (size_t)T_DIM * H_DIM;
    const size_t BH_ELEMS = (size_t)H_DIM * H_DIM;
    const size_t NEED = S_BYTES + 2 * XH_ELEMS * 2 + 2 * BH_ELEMS * 2;  // ~80 MB

    float* s = (float*)d_ws;

    if (ws_size >= NEED) {
        unsigned short* xhi  = (unsigned short*)((char*)d_ws + S_BYTES);
        unsigned short* xlo  = xhi + XH_ELEMS;
        unsigned short* bthi = xlo + XH_ELEMS;
        unsigned short* btlo = bthi + BH_ELEMS;

        split_bf16<<<dim3((unsigned)(XH_ELEMS / 4 / 256)), 256, 0, stream>>>(x, xhi, xlo);
        split_transpose_bf16<<<dim3(H_DIM / 64, H_DIM / 64), 256, 0, stream>>>(b, bthi, btlo);
        ssm_gemm_wmma<<<dim3(H_DIM / 128, T_DIM / 128), 256, 0, stream>>>(
            xhi, xlo, bthi, btlo, s);
    } else {
        ssm_gemm_fused<<<dim3(H_DIM / 64, T_DIM / 64), 128, 0, stream>>>(x, b, s);
    }
    ssm_scan<<<dim3(H_DIM / 256), 256, 0, stream>>>(a, s, out);
}
